// Model_1778116460915
// MI455X (gfx1250) — matrix operand order repaired and checked by reading, not compiled, not run on hardware
//
#include <hip/hip_runtime.h>

// ---------------------------------------------------------------------------
// Types for CDNA5 WMMA (gfx1250, wave32): D = A(16x32 bf16) x B(32x16 bf16) + C(16x16 f32)
// ---------------------------------------------------------------------------
typedef __attribute__((ext_vector_type(16))) __bf16 v16bf;
typedef __attribute__((ext_vector_type(8)))  float  v8f;

// ---------------------------------------------------------------------------
// Packed-weight workspace layout (units: unsigned short / bf16 elements).
// Each B fragment = 32 lanes * 16 bf16 = 512 elems; fragment f of a matrix is
// stored at f*512, lane-major (lane's 16 elems contiguous -> one 32B load).
// Fragment order within a matrix: frag = kt * ntiles + nt.
// ---------------------------------------------------------------------------
constexpr int FRAG      = 512;
constexpr int OFF_EMBED = 0;                       //  1 * (1*4) frags
constexpr int OFF_TC1   = OFF_EMBED +  4 * FRAG;   //  9 * (2*4)
constexpr int OFF_CHEB  = OFF_TC1   + 72 * FRAG;   //  3 * (2*4)
constexpr int OFF_TC2   = OFF_CHEB  + 24 * FRAG;   //  9 * (2*4)
constexpr int OFF_FF1   = OFF_TC2   + 72 * FRAG;   //  3 * (2*16)
constexpr int OFF_FF2   = OFF_FF1   + 96 * FRAG;   //  3 * (8*4)
constexpr int OFF_FS    = OFF_FF2   + 96 * FRAG;   //  3 * (2*4)
constexpr int OFF_FG    = OFF_FS    + 24 * FRAG;   //  3 * (4*4)
constexpr int OFF_OUT   = OFF_FG    + 48 * FRAG;   //  1 * (2*1)
// total = 438 frags = 224256 ushort = 448512 bytes in d_ws

// ---------------------------------------------------------------------------
// Weight packing kernel: W is [Ksrc, Nsrc] row-major f32 (gridDim.y matrices,
// wstride apart). Emits bf16 B fragments in this file's contiguous k order (same as load_afrag):
//   element e of lane L in K-tile kt, N-tile nt maps to
//   K = kt*32 + (L>=16 ? 16 : 0) + e,  N = nt*16 + (L & 15)
// ---------------------------------------------------------------------------
__global__ void pack_w(const float* __restrict__ W, unsigned short* __restrict__ out,
                       int Ksrc, int Nsrc, int ktiles, int ntiles,
                       long wstride, long ostride) {
  int m   = blockIdx.y;
  int tid = blockIdx.x * blockDim.x + threadIdx.x;
  int total = ktiles * ntiles * 32;
  if (tid >= total) return;
  int lane = tid & 31, f = tid >> 5;
  int kt = f / ntiles, nt = f - kt * ntiles;
  const float* w = W + (long)m * wstride;
  __bf16* o = (__bf16*)(out + (long)m * ostride + ((long)f * 32 + lane) * 16);
  int n  = nt * 16 + (lane & 15);
  int kb = kt * 32 + ((lane >> 4) << 4);
#pragma unroll
  for (int e = 0; e < 16; ++e) {
    int k = kb + e;
    float v = (k < Ksrc && n < Nsrc) ? w[(long)k * Nsrc + n] : 0.f;
    o[e] = (__bf16)v;
  }
}

// ---------------------------------------------------------------------------
// A fragment (16x32 bf16) from an LDS row-major tile.
// Lane L holds row M = L&15; K = 2v + (L>=16 ? 16:0) + (e&1).
// ---------------------------------------------------------------------------
__device__ __forceinline__ v16bf load_afrag(const __bf16* rp, int hi) {
  v16bf a;
#pragma unroll
  for (int v = 0; v < 8; ++v) {
    int k = 2 * v + hi * 16;
    a[2 * v]     = rp[k];
    a[2 * v + 1] = rp[k + 1];
  }
  return a;
}

// Load KT A fragments (one per 32-wide K tile) of a staged LDS tile.
template <int KT>
__device__ __forceinline__ void load_aset(const __bf16* tin, int stride, int lane, v16bf* a) {
  int r = lane & 15, hi = lane >> 4;
#pragma unroll
  for (int kt = 0; kt < KT; ++kt) a[kt] = load_afrag(tin + r * stride + kt * 32, hi);
}

// C/D fragment element v of lane L is (M = v + 8*(L>>4), N = L&15) within a 16x16 tile.
// Optionally fuses ReLU into the f32->bf16 conversion at store time.
template <bool RELU>
__device__ __forceinline__ void store_tile4(const v8f* c, __bf16* lds, int stride,
                                            int lane, int col0) {
  int n = lane & 15, hi = lane >> 4;
#pragma unroll
  for (int nt = 0; nt < 4; ++nt)
#pragma unroll
    for (int v = 0; v < 8; ++v) {
      float val = c[nt][v];
      if (RELU) val = val > 0.f ? val : 0.f;
      lds[(v + 8 * hi) * stride + col0 + nt * 16 + n] = (__bf16)val;
    }
}

__device__ __forceinline__ void bias_init4(v8f* c, const float* b, int lane) {
  int n = lane & 15;
#pragma unroll
  for (int nt = 0; nt < 4; ++nt) {
    float bv = b[nt * 16 + n];
#pragma unroll
    for (int v = 0; v < 8; ++v) c[nt][v] = bv;
  }
}

// Tile GEMM on preloaded A fragments:
// C[0..3] += A * B(packed weights, global K-tile kt0+kt, N-tiles nt0..nt0+3)
template <int KT>
__device__ __forceinline__ void mm_pre(const v16bf* a, const unsigned short* wf,
                                       int kt0, int ntot, int nt0, v8f* c, int lane) {
  const v16bf* W = (const v16bf*)wf;
#pragma unroll
  for (int kt = 0; kt < KT; ++kt)
#pragma unroll
    for (int nt = 0; nt < 4; ++nt) {
      v16bf b = W[((kt0 + kt) * ntot + nt0 + nt) * 32 + lane];
      c[nt] = __builtin_amdgcn_wmma_f32_16x16x32_bf16(false, a[kt], false, b, (short)0,
                                                      c[nt], false, false);
    }
}

// Convenience: load A set then GEMM (for single-use tiles).
template <int KT>
__device__ __forceinline__ void mm_acc(const __bf16* tin, int stride,
                                       const unsigned short* wf, int kt0, int ntot, int nt0,
                                       v8f* c, int lane) {
  v16bf a[KT];
  load_aset<KT>(tin, stride, lane, a);
  mm_pre<KT>(a, wf, kt0, ntot, nt0, c, lane);
}

// Row reduction across the 16-lane half holding this row (xor masks < 16 stay in-half).
__device__ __forceinline__ float rsum16(float x) {
#pragma unroll
  for (int m = 1; m < 16; m <<= 1) x += __shfl_xor(x, m, 32);
  return x;
}

__device__ __forceinline__ void lnorm(v8f* c, const float* g, const float* b, int lane) {
  int n = lane & 15;
  float gv[4], bv[4];
#pragma unroll
  for (int nt = 0; nt < 4; ++nt) { gv[nt] = g[nt * 16 + n]; bv[nt] = b[nt * 16 + n]; }
#pragma unroll
  for (int v = 0; v < 8; ++v) {
    float s  = c[0][v] + c[1][v] + c[2][v] + c[3][v];
    float mu = rsum16(s) * (1.f / 64.f);
    float d0 = c[0][v] - mu, d1 = c[1][v] - mu, d2 = c[2][v] - mu, d3 = c[3][v] - mu;
    float vr = rsum16(d0 * d0 + d1 * d1 + d2 * d2 + d3 * d3) * (1.f / 64.f);
    float ri = rsqrtf(vr + 1e-5f);
    c[0][v] = d0 * ri * gv[0] + bv[0];
    c[1][v] = d1 * ri * gv[1] + bv[1];
    c[2][v] = d2 * ri * gv[2] + bv[2];
    c[3][v] = d3 * ri * gv[3] + bv[3];
  }
}

// TemporalConv(kernel_size=1): P * sigmoid(Q) + R  (final ReLU is fused into the
// caller's tile store). A fragments of the input tile are loaded once and shared
// by all three GEMMs. Register-lean order keeps only 2 accumulator sets live.
__device__ __forceinline__ void tconv(const __bf16* tin, int stride,
                                      const unsigned short* wf, int matstride,
                                      const float* bias, v8f* o, int lane) {
  v16bf a[2];
  load_aset<2>(tin, stride, lane, a);
  v8f Q[4], P[4];
  bias_init4(Q, bias + 64, lane);
  mm_pre<2>(a, wf + matstride, 0, 4, 0, Q, lane);
#pragma unroll
  for (int nt = 0; nt < 4; ++nt)
#pragma unroll
    for (int v = 0; v < 8; ++v) Q[nt][v] = 1.f / (1.f + __expf(-Q[nt][v]));
  bias_init4(P, bias, lane);
  mm_pre<2>(a, wf, 0, 4, 0, P, lane);
  int n = lane & 15;
#pragma unroll
  for (int nt = 0; nt < 4; ++nt) {
    float br = bias[128 + nt * 16 + n];
#pragma unroll
    for (int v = 0; v < 8; ++v) o[nt][v] = P[nt][v] * Q[nt][v] + br;
  }
  mm_pre<2>(a, wf + 2 * matstride, 0, 4, 0, o, lane);
}

// ---------------------------------------------------------------------------
// Fused STGCN kernel: ONE wave32 per workgroup (barriers are free S_NOPs and
// act only as compiler fences for the cross-lane LDS relayout). Each wave owns
// 16 node rows as WMMA C fragments; all matmuls via v_wmma_f32_16x16x32_bf16.
// ---------------------------------------------------------------------------
__global__ __launch_bounds__(32) void st_fused(
    const float* __restrict__ x, const unsigned short* __restrict__ wpk,
    const float* eb,  const float* tc1b, const float* chebb, const float* tc2b,
    const float* ln1g, const float* ln1b, const float* ln2g, const float* ln2b,
    const float* ffb1, const float* ffb2, const float* fsb,  const float* fgb,
    const float* blkg, const float* blkb, const float* outb,
    float* __restrict__ outp, float* __restrict__ xep, int Nn) {
  constexpr int S0 = 136;  // wide tile stride (holds 16x128 xg), bank-skewed
  constexpr int S1 = 72;   // narrow tile stride (16x64)
  __shared__ __align__(16) __bf16 t0[16 * S0];
  __shared__ __align__(16) __bf16 t1[16 * S1];

  const int lane = threadIdx.x & 31;
  const int ln15 = lane & 15;
  const int hi   = lane >> 4;
  const int node0 = blockIdx.x * 16;
  const bool full = (node0 + 16 <= Nn);  // wave-uniform: hot path has no guards

  __builtin_prefetch(wpk + OFF_EMBED, 0, 3);

  // ---- stage x tile (16 rows x 16 feats), zero-pad K to 32 for the WMMA ----
  {
    int row = node0 + ln15;
    if (full) {
#pragma unroll
      for (int cc = 0; cc < 8; ++cc) {
        int c = hi * 8 + cc;
        t0[ln15 * S0 + c]      = (__bf16)x[(long)row * 16 + c];
        t0[ln15 * S0 + 16 + c] = (__bf16)0.f;
      }
    } else {
#pragma unroll
      for (int cc = 0; cc < 8; ++cc) {
        int c = hi * 8 + cc;
        float v = (row < Nn) ? x[(long)row * 16 + c] : 0.f;
        t0[ln15 * S0 + c]      = (__bf16)v;
        t0[ln15 * S0 + 16 + c] = (__bf16)0.f;
      }
    }
  }
  __syncthreads();

  // ---- embed: h = x @ embed_w + embed_b ----
  v8f q[4];
  bias_init4(q, eb, lane);
  mm_acc<1>(t0, S0, wpk + OFF_EMBED, 0, 4, 0, q, lane);

  // x_embed output (second tuple element)
  if (full) {
    float* xb = xep + (long)node0 * 64;
#pragma unroll
    for (int nt = 0; nt < 4; ++nt)
#pragma unroll
      for (int v = 0; v < 8; ++v)
        xb[(long)(v + 8 * hi) * 64 + nt * 16 + ln15] = q[nt][v];
  } else {
#pragma unroll
    for (int nt = 0; nt < 4; ++nt)
#pragma unroll
      for (int v = 0; v < 8; ++v) {
        int node = node0 + v + 8 * hi;
        if (node < Nn) xep[(long)node * 64 + nt * 16 + ln15] = q[nt][v];
      }
  }

  // ---- 3 ST-transformer layers ----
  for (int i = 0; i < 3; ++i) {
    // STConv: t = tconv1(q); u = relu(t @ cheb + b); o = tconv2(u)
    __syncthreads();
    store_tile4<false>(q, t0, S0, lane, 0);
    __syncthreads();
    v8f t[4];
    tconv(t0, S0, wpk + OFF_TC1 + (size_t)(i * 3) * 8 * FRAG, 8 * FRAG,
          tc1b + (i * 3) * 64, t, lane);

    __syncthreads();
    store_tile4<true>(t, t0, S0, lane, 0);  // ReLU fused (TemporalConv output)
    __syncthreads();
    v8f u[4];
    bias_init4(u, chebb + i * 64, lane);
    mm_acc<2>(t0, S0, wpk + OFF_CHEB + (size_t)i * 8 * FRAG, 0, 4, 0, u, lane);

    __syncthreads();
    store_tile4<true>(u, t0, S0, lane, 0);  // ReLU fused (ChebConv + relu)
    __syncthreads();
    v8f o4[4];
    tconv(t0, S0, wpk + OFF_TC2 + (size_t)(i * 3) * 8 * FRAG, 8 * FRAG,
          tc2b + (i * 3) * 64, o4, lane);

    // fgx = [q, relu(o)] @ fg_w + fg_b   (K = 128)
    __syncthreads();
    store_tile4<false>(q, t0, S0, lane, 0);
    store_tile4<true>(o4, t0, S0, lane, 64);  // ReLU fused (TemporalConv output)
    __syncthreads();
    v8f fgx[4];
    bias_init4(fgx, fgb + i * 64, lane);
    mm_acc<4>(t0, S0, wpk + OFF_FG + (size_t)i * 16 * FRAG, 0, 4, 0, fgx, lane);

    // xn = LN1(q); stage it in t1 (kept across FF chunks)
    {
      v8f xn[4];
#pragma unroll
      for (int nt = 0; nt < 4; ++nt) xn[nt] = q[nt];
      lnorm(xn, ln1g + i * 64, ln1b + i * 64, lane);
      __syncthreads();
      store_tile4<false>(xn, t1, S1, lane, 0);
      __syncthreads();
    }

    // ff = relu(xn @ W1 + b1) @ W2 + b2, 4 chunks of 64 hidden cols.
    // A fragments of xn are loaded once and reused by all 4 chunk GEMMs.
    v8f ffa[4];
    bias_init4(ffa, ffb2 + i * 64, lane);
    {
      v16bf axn[2];
      load_aset<2>(t1, S1, lane, axn);
      for (int c = 0; c < 4; ++c) {
        v8f hc[4];
        bias_init4(hc, ffb1 + i * 256 + c * 64, lane);
        mm_pre<2>(axn, wpk + OFF_FF1 + (size_t)i * 32 * FRAG, 0, 16, 4 * c, hc, lane);
        __syncthreads();
        store_tile4<true>(hc, t0, S0, lane, 0);  // ReLU fused (FF hidden)
        __syncthreads();
        mm_acc<2>(t0, S0, wpk + OFF_FF2 + (size_t)i * 32 * FRAG, 2 * c, 4, 0, ffa, lane);
      }
    }

    // us = LN2(ff + xn)  (xn re-read from LDS, bf16)
    v8f us[4];
#pragma unroll
    for (int nt = 0; nt < 4; ++nt)
#pragma unroll
      for (int v = 0; v < 8; ++v)
        us[nt][v] = ffa[nt][v] + (float)t1[(v + 8 * hi) * S1 + nt * 16 + ln15];
    lnorm(us, ln2g + i * 64, ln2b + i * 64, lane);

    // g = sigmoid(us @ fs_w + fs_b + fgx);  out_s = g*us + (1-g)*fgx
    __syncthreads();
    store_tile4<false>(us, t1, S1, lane, 0);
    __syncthreads();
    v8f gg[4];
    bias_init4(gg, fsb + i * 64, lane);
    mm_acc<2>(t1, S1, wpk + OFF_FS + (size_t)i * 8 * FRAG, 0, 4, 0, gg, lane);

#pragma unroll
    for (int nt = 0; nt < 4; ++nt)
#pragma unroll
      for (int v = 0; v < 8; ++v) {
        float g  = 1.f / (1.f + __expf(-(gg[nt][v] + fgx[nt][v])));
        float os = g * us[nt][v] + (1.f - g) * fgx[nt][v];
        q[nt][v] = os + q[nt][v];  // residual
      }
    lnorm(q, blkg + i * 64, blkb + i * 64, lane);  // h for next layer
  }

  // ---- out = h @ out_w + out_b  (N padded 12 -> 16, single N-tile) ----
  __syncthreads();
  store_tile4<false>(q, t1, S1, lane, 0);
  __syncthreads();
  {
    v8f co;
    float bv = (ln15 < 12) ? outb[ln15] : 0.f;
#pragma unroll
    for (int v = 0; v < 8; ++v) co[v] = bv;
    const v16bf* W = (const v16bf*)(wpk + OFF_OUT);
#pragma unroll
    for (int kt = 0; kt < 2; ++kt) {
      v16bf a = load_afrag(t1 + ln15 * S1 + kt * 32, hi);
      v16bf b = W[kt * 32 + lane];
      co = __builtin_amdgcn_wmma_f32_16x16x32_bf16(false, a, false, b, (short)0,
                                                   co, false, false);
    }
    if (full) {
      if (ln15 < 12) {
        float* ob = outp + (long)node0 * 12 + ln15;
#pragma unroll
        for (int v = 0; v < 8; ++v) ob[(long)(v + 8 * hi) * 12] = co[v];
      }
    } else {
#pragma unroll
      for (int v = 0; v < 8; ++v) {
        int node = node0 + v + 8 * hi;
        if (node < Nn && ln15 < 12) outp[(long)node * 12 + ln15] = co[v];
      }
    }
  }
}

// ---------------------------------------------------------------------------
// Launcher
// ---------------------------------------------------------------------------
extern "C" void kernel_launch(void* const* d_in, const int* in_sizes, int n_in,
                              void* d_out, int out_size, void* d_ws, size_t ws_size,
                              hipStream_t stream) {
  const float* x = (const float*)d_in[0];
  // d_in[1] edge_index, d_in[2] edge_weight: mathematically unused (ChebConv K=1)
  unsigned short* wp = (unsigned short*)d_ws;
  const int Nn = in_sizes[0] / 16;

#define PACK(WPTR, OFF, KS, NS, KT, NT, NM, WSTR)                                    \
  {                                                                                  \
    int tot = (KT) * (NT) * 32;                                                      \
    dim3 g((tot + 255) / 256, (NM));                                                 \
    pack_w<<<g, 256, 0, stream>>>((const float*)(WPTR), wp + (OFF), (KS), (NS),      \
                                  (KT), (NT), (long)(WSTR), (long)((KT) * (NT) * FRAG)); \
  }
  PACK(d_in[3],  OFF_EMBED, 16,  64, 1, 4,  1, 0);      // embed_w [16,64]
  PACK(d_in[5],  OFF_TC1,   64,  64, 2, 4,  9, 4096);   // tc1_w  [3,3,64,64]
  PACK(d_in[7],  OFF_CHEB,  64,  64, 2, 4,  3, 4096);   // cheb_w [3,64,64]
  PACK(d_in[9],  OFF_TC2,   64,  64, 2, 4,  9, 4096);   // tc2_w  [3,3,64,64]
  PACK(d_in[15], OFF_FF1,   64, 256, 2, 16, 3, 16384);  // ff_w1  [3,64,256]
  PACK(d_in[17], OFF_FF2,  256,  64, 8, 4,  3, 16384);  // ff_w2  [3,256,64]
  PACK(d_in[19], OFF_FS,    64,  64, 2, 4,  3, 4096);   // fs_w   [3,64,64]
  PACK(d_in[21], OFF_FG,   128,  64, 4, 4,  3, 8192);   // fg_w   [3,128,64]
  PACK(d_in[25], OFF_OUT,   64,  12, 2, 1,  1, 0);      // out_w  [64,12]
#undef PACK

  float* outp = (float*)d_out;
  float* xep  = outp + (size_t)Nn * 12;
  int blocks  = (Nn + 15) / 16;
  st_fused<<<blocks, 32, 0, stream>>>(
      x, wp,
      (const float*)d_in[4],  (const float*)d_in[6],  (const float*)d_in[8],
      (const float*)d_in[10], (const float*)d_in[11], (const float*)d_in[12],
      (const float*)d_in[13], (const float*)d_in[14], (const float*)d_in[16],
      (const float*)d_in[18], (const float*)d_in[20], (const float*)d_in[22],
      (const float*)d_in[23], (const float*)d_in[24], (const float*)d_in[26],
      outp, xep, Nn);
}
